// GraphFilter_15788299780743
// MI455X (gfx1250) — matrix operand order repaired and checked by reading, not compiled, not run on hardware
//
#include <hip/hip_runtime.h>
#include <math.h>

#define BB 8
#define HH 8
#define LL 512
#define DD 512
#define DH 64
#define BHN (BB*HH)
#define KSM 256
#define TOPP 0.5f
#define EPSV 1e-10f

typedef __attribute__((ext_vector_type(16))) __bf16 v16bf;
typedef __attribute__((ext_vector_type(8)))  float  v8f;

// ---- helpers -------------------------------------------------------------

static __device__ __forceinline__ float wave_sum(float v) {
  #pragma unroll
  for (int o = 16; o > 0; o >>= 1) v += __shfl_down(v, o, 32);
  return __shfl(v, 0, 32);
}
static __device__ __forceinline__ int wave_sum_i(int v) {
  #pragma unroll
  for (int o = 16; o > 0; o >>= 1) v += __shfl_down(v, o, 32);
  return __shfl(v, 0, 32);
}
static __device__ __forceinline__ float wave_max(float v) {
  #pragma unroll
  for (int o = 16; o > 0; o >>= 1) v = fmaxf(v, __shfl_down(v, o, 32));
  return __shfl(v, 0, 32);
}
// A-matrix (16x32 bf16): K index for half j, lane group grp (ISA 7.12.2)
static __device__ __forceinline__ int a_kidx(int j, int grp) {
  int v = j >> 1, p = j & 1;
  return ((v & 3) * 2 + p) + 8 * grp + 16 * (v >> 2);
}
// inverse of a_kidx: kd (0..31) -> lane group g2 and half index j2
static __device__ __forceinline__ void a_inv(int kd, int& g2, int& j2) {
  g2 = (kd & 15) >> 3;
  int kdp = kd - 8 * g2;        // in [0..7] or [16..23]
  int v = ((kdp >> 4) << 2) + ((kdp & 7) >> 1);
  j2 = v * 2 + (kdp & 1);
}
static __device__ __forceinline__ float key2f(unsigned kq) {
  unsigned u = (kq & 0x80000000u) ? (kq & 0x7FFFFFFFu) : ~kq;
  return __uint_as_float(u);
}

// ---- K0: zero accumulators ----------------------------------------------
__global__ void gf_init(float* imp, float* dynacc) {
  int t = blockIdx.x * blockDim.x + threadIdx.x;
  if (t < LL * 3) imp[t] = 0.0f;
  if (t == 0) dynacc[0] = 0.0f;
}

// ---- K1: q,k projections (WMMA bf16, K=64); outputs pre-swizzled ---------
// q tiles  : A-layout, buf[((bh*32+it)*2+kc)*512 + lane*16 + j]
// k tiles  : B-layout, buf[((bh*32+jt)*2+kc)*512 + lane*16 + j]
// Two passes (W1->q, W2->k) to keep the live fragment set small (no spills).
__global__ void gf_proj_qk(const float* __restrict__ x,
                           const float* __restrict__ W1, const float* __restrict__ b1,
                           const float* __restrict__ W2, const float* __restrict__ b2,
                           __bf16* __restrict__ qo, __bf16* __restrict__ ko) {
  int lane = threadIdx.x;
  int m = lane & 15, grp = lane >> 4;
  int itile = blockIdx.x * 4 + threadIdx.y;   // 0..31
  int bh = blockIdx.y;                        // 0..63
  int b = bh / HH, h = bh % HH;
  int i0 = itile * 16;

  const float* xrow = x + ((size_t)(b * LL + i0 + m)) * DD + h * DH;
  v16bf a0, a1;
  #pragma unroll
  for (int j = 0; j < 16; ++j) {
    int kd = a_kidx(j, grp);
    a0[j] = (__bf16)xrow[kd];
    a1[j] = (__bf16)xrow[32 + kd];
  }
  #pragma unroll
  for (int sel = 0; sel < 2; ++sel) {
    const float* W  = sel ? W2 : W1;
    const float* bi = sel ? b2 : b1;
    __bf16*      oo = sel ? ko : qo;
    #pragma unroll
    for (int nt = 0; nt < 4; ++nt) {
      int n0 = nt * 16;
      const float* wr = W + (size_t)(n0 + m) * DH;
      v16bf w0, w1;
      #pragma unroll
      for (int j = 0; j < 16; ++j) {
        int kd = a_kidx(j, grp);
        w0[j] = (__bf16)wr[kd];
        w1[j] = (__bf16)wr[32 + kd];
      }
      v8f acc = {};
      acc = __builtin_amdgcn_wmma_f32_16x16x32_bf16(false, a0, false, w0, (short)0, acc, false, false);
      acc = __builtin_amdgcn_wmma_f32_16x16x32_bf16(false, a1, false, w1, (short)0, acc, false, false);
      float bias = bi[n0 + m];
      // target swizzle coords: d = n0+m is the K position of q/k feature dim
      int d = n0 + m, kc = d >> 5, kd2 = d & 31;
      int gt, jt;
      if (sel == 0) { a_inv(kd2, gt, jt); }            // q consumed as A
      else          { a_inv(kd2, gt, jt); }            // k consumed as B
      size_t tbase = (((size_t)bh * 32 + itile) * 2 + kc) * 512;
      #pragma unroll
      for (int r = 0; r < 8; ++r) {
        int m2 = r + 8 * grp;                          // row within 16-tile
        oo[tbase + (size_t)(m2 + 16 * gt) * 16 + jt] = (__bf16)(acc[r] + bias);
      }
    }
  }
}

// ---- K2: xp = x @ Wp^T + bp (WMMA bf16, K=512); output pre-swizzled ------
// xp tiles : B-layout for gf_out, buf[(((b*8+h)*4+nt)*16+kc)*512 + lane*16 + j]
__global__ void gf_proj_xp(const float* __restrict__ x, const float* __restrict__ Wp,
                           const float* __restrict__ bp, __bf16* __restrict__ xpo) {
  int lane = threadIdx.x;
  int m = lane & 15, grp = lane >> 4;
  int itile = blockIdx.x * 4 + threadIdx.y;   // 0..31
  int ntile = blockIdx.y;                     // 0..31
  int b = blockIdx.z;                         // 0..7
  int i0 = itile * 16, n0 = ntile * 16;
  const float* xrow = x + ((size_t)(b * LL + i0 + m)) * DD;
  const float* wrow = Wp + (size_t)(n0 + m) * DD;
  v8f acc = {};
  for (int k0 = 0; k0 < DD; k0 += 32) {
    if (k0 + 32 < DD) {
      __builtin_prefetch(xrow + k0 + 32, 0, 0);
      __builtin_prefetch(wrow + k0 + 32, 0, 0);
    }
    v16bf a, bm;
    #pragma unroll
    for (int j = 0; j < 16; ++j) {
      a[j]  = (__bf16)xrow[k0 + a_kidx(j, grp)];
      bm[j] = (__bf16)wrow[k0 + a_kidx(j, grp)];
    }
    acc = __builtin_amdgcn_wmma_f32_16x16x32_bf16(false, a, false, bm, (short)0, acc, false, false);
  }
  float bias = bp[n0 + m];
  int c = n0 + m;                  // output feature (0..511)
  int h2 = c >> 6, nt = (c >> 4) & 3, ncol = c & 15;
  size_t obase = ((((size_t)b * HH + h2) * 4 + nt) * 16) * 512;
  #pragma unroll
  for (int r = 0; r < 8; ++r) {
    int jrow = i0 + r + 8 * grp;   // K position (row of xp) for gf_out
    int kc = jrow >> 5, kd = jrow & 31;
    int gb, jb; a_inv(kd, gb, jb);
    xpo[obase + (size_t)kc * 512 + (size_t)(ncol + 16 * gb) * 16 + jb] = (__bf16)(acc[r] + bias);
  }
}

// ---- K3: adj = gelu(q @ k^T); vectorized fragment loads ------------------
__global__ void gf_attn(const __bf16* __restrict__ q,
                        const __bf16* __restrict__ k,
                        float* __restrict__ adj) {
  int lane = threadIdx.x;
  int m = lane & 15, grp = lane >> 4;
  int itile = blockIdx.x * 4 + threadIdx.y;   // 0..31
  int jtile = blockIdx.y;                     // 0..31
  int bh = blockIdx.z;                        // 0..63
  int i0 = itile * 16, j0 = jtile * 16;
  const __bf16* qt = q + (((size_t)bh * 32 + itile) * 2) * 512 + (size_t)lane * 16;
  const __bf16* kt = k + (((size_t)bh * 32 + jtile) * 2) * 512 + (size_t)lane * 16;
  v16bf a0 = *(const v16bf*)(qt);
  v16bf a1 = *(const v16bf*)(qt + 512);
  v16bf b0 = *(const v16bf*)(kt);
  v16bf b1 = *(const v16bf*)(kt + 512);
  v8f c = {};
  c = __builtin_amdgcn_wmma_f32_16x16x32_bf16(false, a0, false, b0, (short)0, c, false, false);
  c = __builtin_amdgcn_wmma_f32_16x16x32_bf16(false, a1, false, b1, (short)0, c, false, false);
  #pragma unroll
  for (int r = 0; r < 8; ++r) {
    float v = c[r];
    v = 0.5f * v * (1.0f + erff(v * 0.70710678118654752f));   // exact GELU
    adj[((size_t)bh * LL + i0 + r + 8 * grp) * LL + j0 + m] = v;
  }
}

// ---- K4: exact 256th-order-stat threshold (radix select) + gating --------
__global__ void gf_select(float* __restrict__ adj, const float* __restrict__ Wg,
                          float* __restrict__ gates, float* __restrict__ imp,
                          float* __restrict__ dynacc) {
  int lane = threadIdx.x;
  int row = blockIdx.x * blockDim.y + threadIdx.y;     // bh*L + i
  int i = row % LL;
  float* arow = adj + (size_t)row * LL;

  float vals[16]; unsigned keys[16];
  #pragma unroll
  for (int t = 0; t < 16; ++t) {
    float f = arow[lane + 32 * t];
    vals[t] = f;
    unsigned u = __float_as_uint(f);
    keys[t] = (u & 0x80000000u) ? ~u : (u | 0x80000000u);  // monotonic key
  }
  unsigned prefix = 0; int kk = KSM - 1;
  for (int bit = 31; bit >= 0; --bit) {
    unsigned above = (bit == 31) ? 0u : (0xFFFFFFFFu << (bit + 1));
    int cnt = 0;
    #pragma unroll
    for (int t = 0; t < 16; ++t)
      if (((keys[t] & above) == prefix) && !((keys[t] >> bit) & 1u)) cnt++;
    cnt = wave_sum_i(cnt);
    if (kk >= cnt) { kk -= cnt; prefix |= (1u << bit); }
  }
  float thr = key2f(prefix);

  float g0 = 0.f, g1 = 0.f, g2 = 0.f;
  #pragma unroll
  for (int t = 0; t < 16; ++t) {
    int j = lane + 32 * t;
    float f = vals[t];
    f = (f > thr) ? f : 0.0f;
    vals[t] = f;
    g0 += f * Wg[j]; g1 += f * Wg[LL + j]; g2 += f * Wg[2 * LL + j];
  }
  #pragma unroll
  for (int t = 0; t < 16; ++t) arow[lane + 32 * t] = vals[t];
  g0 = wave_sum(g0); g1 = wave_sum(g1); g2 = wave_sum(g2);

  if (lane == 0) {
    float mx = fmaxf(g0, fmaxf(g1, g2));
    float e0 = expf(g0 - mx), e1 = expf(g1 - mx), e2 = expf(g2 - mx);
    float s = e0 + e1 + e2;
    float p[3] = { e0 / s, e1 / s, e2 / s };
    float ent = -(p[0] * logf(p[0] + EPSV) + p[1] * logf(p[1] + EPSV) + p[2] * logf(p[2] + EPSV));
    atomicAdd(dynacc, ent);
    int ord[3] = {0, 1, 2};
    #pragma unroll
    for (int a2 = 1; a2 < 3; ++a2) {
      int o = ord[a2]; int j2 = a2;
      while (j2 > 0 && p[o] > p[ord[j2 - 1]]) { ord[j2] = ord[j2 - 1]; --j2; }
      ord[j2] = o;
    }
    float sp[3] = { p[ord[0]], p[ord[1]], p[ord[2]] };
    float cum = 0.f; bool mm[3];
    #pragma unroll
    for (int t = 0; t < 3; ++t) { cum += sp[t]; mm[t] = (cum > TOPP); }
    int thr_idx = mm[0] ? 0 : (mm[1] ? 1 : (mm[2] ? 2 : 0));
    mm[thr_idx] = false;
    float go[3] = {0.f, 0.f, 0.f};
    #pragma unroll
    for (int t = 0; t < 3; ++t) {
      float sel = mm[t] ? 0.0f : 1.0f;
      go[ord[t]] = sel;
      atomicAdd(&imp[i * 3 + t], sel * sp[t]);
    }
    gates[(size_t)row * 3 + 0] = go[0];
    gates[(size_t)row * 3 + 1] = go[1];
    gates[(size_t)row * 3 + 2] = go[2];
  }
}

// ---- K5: mask + row softmax; writes A-layout swizzled bf16 ---------------
// adjf tiles: buf[((bh*32+it)*16+kc)*512 + lane*16 + j]
__global__ void gf_masksoft(const float* __restrict__ adj, const float* __restrict__ gates,
                            const float* __restrict__ masks, __bf16* __restrict__ adjf) {
  int lane = threadIdx.x;
  int row = blockIdx.x * blockDim.y + threadIdx.y;     // bh*L + i
  int bh = row / LL, i = row % LL;
  float g0 = gates[(size_t)row * 3 + 0];
  float g1 = gates[(size_t)row * 3 + 1];
  float g2 = gates[(size_t)row * 3 + 2];
  const float* m0 = masks + ((size_t)i * 3 + 0) * LL;
  const float* m1 = masks + ((size_t)i * 3 + 1) * LL;
  const float* m2 = masks + ((size_t)i * 3 + 2) * LL;
  const float* arow = adj + (size_t)row * LL;
  float v[16]; float mx = -3.4028235e38f;
  #pragma unroll
  for (int t = 0; t < 16; ++t) {
    int j = lane + 32 * t;
    float mk = g0 * m0[j] + g1 * m1[j] + g2 * m2[j] + ((j == i) ? 1.0f : 0.0f);
    float a = arow[j] * mk;
    v[t] = a; mx = fmaxf(mx, a);
  }
  mx = wave_max(mx);
  float s = 0.f;
  #pragma unroll
  for (int t = 0; t < 16; ++t) { float e = expf(v[t] - mx); v[t] = e; s += e; }
  s = wave_sum(s);
  float inv = 1.0f / fmaxf(s, 1e-12f);   // softmax; abs-renorm is identity
  int it = i >> 4, mrow = i & 15;
  __bf16* obase = adjf + (((size_t)bh * 32 + it) * 16) * 512;
  #pragma unroll
  for (int t = 0; t < 16; ++t) {
    int jj = lane + 32 * t;
    int kc = jj >> 5, kd = jj & 31;
    int ga, ja; a_inv(kd, ga, ja);
    obase[(size_t)kc * 512 + (size_t)(mrow + 16 * ga) * 16 + ja] = (__bf16)(v[t] * inv);
  }
}

// ---- K6: out = adj_final @ xp; fully vectorized fragment loads -----------
__global__ void gf_out(const __bf16* __restrict__ adjf,
                       const __bf16* __restrict__ xp,
                       float* __restrict__ out) {
  int lane = threadIdx.x;
  int m = lane & 15, grp = lane >> 4;
  int itile = blockIdx.x;        // 0..31
  int ntile = threadIdx.y;       // 0..3
  int bh = blockIdx.z;           // 0..63
  int b = bh / HH, h = bh % HH;
  int i0 = itile * 16, n0 = ntile * 16;
  const __bf16* abase = adjf + (((size_t)bh * 32 + itile) * 16) * 512 + (size_t)lane * 16;
  const __bf16* bbase = xp + ((((size_t)b * HH + h) * 4 + ntile) * 16) * 512 + (size_t)lane * 16;
  v8f c = {};
  #pragma unroll
  for (int kc = 0; kc < 16; ++kc) {
    if (kc + 1 < 16) {
      __builtin_prefetch(abase + (kc + 1) * 512, 0, 0);
      __builtin_prefetch(bbase + (kc + 1) * 512, 0, 0);
    }
    v16bf a  = *(const v16bf*)(abase + kc * 512);
    v16bf bm = *(const v16bf*)(bbase + kc * 512);
    c = __builtin_amdgcn_wmma_f32_16x16x32_bf16(false, a, false, bm, (short)0, c, false, false);
  }
  #pragma unroll
  for (int r = 0; r < 8; ++r)
    out[((size_t)(b * LL + i0 + r + 8 * grp)) * DD + h * DH + n0 + m] = c[r];
}

// ---- K7: final loss ------------------------------------------------------
__global__ void gf_loss(const float* __restrict__ imp, const float* __restrict__ dynacc,
                        float* __restrict__ out_loss) {
  int lane = threadIdx.x;
  float s = 0.f, sq = 0.f;
  for (int t = lane; t < LL * 3; t += 32) { float v = imp[t]; s += v; sq += v * v; }
  s = wave_sum(s); sq = wave_sum(sq);
  if (lane == 0) {
    const float N = (float)(LL * 3);
    float mean = s / N;
    float var = (sq - N * mean * mean) / (N - 1.0f);   // ddof=1
    float loss_imp = var / (mean * mean + EPSV);
    float loss_dyn = dynacc[0] / (float)(BHN * 3);
    out_loss[0] = loss_imp + 0.1f * loss_dyn;
  }
}

// ---- host ----------------------------------------------------------------
extern "C" void kernel_launch(void* const* d_in, const int* in_sizes, int n_in,
                              void* d_out, int out_size, void* d_ws, size_t ws_size,
                              hipStream_t stream) {
  const float* x     = (const float*)d_in[0];
  const float* masks = (const float*)d_in[1];
  const float* W1    = (const float*)d_in[2];
  const float* b1    = (const float*)d_in[3];
  const float* W2    = (const float*)d_in[4];
  const float* b2    = (const float*)d_in[5];
  const float* Wg    = (const float*)d_in[6];
  const float* Wp    = (const float*)d_in[7];
  const float* bp    = (const float*)d_in[8];
  float* out = (float*)d_out;

  char* ws = (char*)d_ws;
  const size_t MB = 1024 * 1024;
  __bf16* qb    = (__bf16*)(ws + 0);            //  4 MiB, swizzled A tiles
  __bf16* kb    = (__bf16*)(ws + 4 * MB);       //  4 MiB, swizzled B tiles
  __bf16* xpb   = (__bf16*)(ws + 8 * MB);       //  4 MiB, swizzled B tiles
  float*  adj   = (float*)(ws + 12 * MB);       // 64 MiB row-major
  __bf16* adjf  = (__bf16*)(ws + 76 * MB);      // 32 MiB, swizzled A tiles
  float*  gates = (float*)(ws + 108 * MB);      // 384 KiB
  float*  imp   = (float*)(ws + 108 * MB + 512 * 1024);
  float*  dyn   = (float*)(ws + 108 * MB + 512 * 1024 + 8192);

  gf_init<<<dim3((LL * 3 + 255) / 256), dim3(256), 0, stream>>>(imp, dyn);
  gf_proj_qk<<<dim3(8, BHN), dim3(32, 4), 0, stream>>>(x, W1, b1, W2, b2, qb, kb);
  gf_proj_xp<<<dim3(8, 32, BB), dim3(32, 4), 0, stream>>>(x, Wp, bp, xpb);
  gf_attn<<<dim3(8, 32, BHN), dim3(32, 4), 0, stream>>>(qb, kb, adj);
  gf_select<<<dim3(BHN * LL / 8), dim3(32, 8), 0, stream>>>(adj, Wg, gates, imp, dyn);
  gf_masksoft<<<dim3(BHN * LL / 8), dim3(32, 8), 0, stream>>>(adj, gates, masks, adjf);
  gf_out<<<dim3(32, 1, BHN), dim3(32, 4), 0, stream>>>(adjf, xpb, out);
  gf_loss<<<dim3(1), dim3(32), 0, stream>>>(imp, dyn, out + (size_t)BB * LL * DD);
}
